// LightweightTemporalAttention_6597069766967
// MI455X (gfx1250) — matrix operand order repaired and checked by reading, not compiled, not run on hardware
//
#include <hip/hip_runtime.h>
#include <hip/hip_bf16.h>

#define HIDDEN 1024
#define NHEAD 8
#define HD 128
#define B_SZ 32
#define T_SZ 9
#define SPATIAL 576   // 18*32

typedef __attribute__((ext_vector_type(16))) __bf16 v16bf;
typedef __attribute__((ext_vector_type(8)))  float  v8f;
typedef __attribute__((ext_vector_type(4)))  float  fvec4;   // clang vector type:
                                                             // legal for nontemporal builtins

__device__ __forceinline__ unsigned short f2bf(float f) {
    unsigned int u = __float_as_uint(f);
    u += 0x7FFFu + ((u >> 16) & 1u);   // round-to-nearest-even
    return (unsigned short)(u >> 16);
}

// ---------------------------------------------------------------------------
// 1) Spatial pooling: one wave32 per (b,t,c). 576 contiguous floats per wave,
//    read as 16B vectors with NT temporal hint (read-once 680 MB stream:
//    don't sweep the 192 MB L2 that the GEMM working set wants), then
//    shuffle-reduced across the wave. Output: bf16 context row-major [B*T, C].
// ---------------------------------------------------------------------------
__global__ void pool_kernel(const float* __restrict__ hc,
                            const float* __restrict__ hb,
                            unsigned short* __restrict__ ctx) {
    int wave = (blockIdx.x * blockDim.x + threadIdx.x) >> 5;   // 0..294911
    int lane = threadIdx.x & 31;
    int c = wave & (HIDDEN - 1);
    int m = wave >> 10;                 // b*T + t
    int b = m / T_SZ, t = m - b * T_SZ;
    const float* src = (t < T_SZ - 1)
        ? hb + (((size_t)t * B_SZ + b) * HIDDEN + c) * SPATIAL
        : hc + ((size_t)b * HIDDEN + c) * SPATIAL;
    const fvec4* s4 = (const fvec4*)src;   // 144 x 16B
    float sum = 0.f;
#pragma unroll
    for (int i = 0; i < 4; ++i) {
        fvec4 v = __builtin_nontemporal_load(s4 + i * 32 + lane);
        sum += v.x + v.y + v.z + v.w;
    }
    if (lane < 16) {
        fvec4 v = __builtin_nontemporal_load(s4 + 128 + lane);
        sum += v.x + v.y + v.z + v.w;
    }
#pragma unroll
    for (int off = 16; off; off >>= 1) sum += __shfl_xor(sum, off, 32);
    if (lane == 0) ctx[wave] = f2bf(sum * (1.0f / (float)SPATIAL));
}

// ---------------------------------------------------------------------------
// 2) f32 -> bf16 conversion (weights; small, L2-resident afterwards)
// ---------------------------------------------------------------------------
__global__ void convert_kernel(const float* __restrict__ in,
                               unsigned short* __restrict__ outp, int n) {
    int i = blockIdx.x * blockDim.x + threadIdx.x;
    if (i < n) outp[i] = f2bf(in[i]);
}

// ---------------------------------------------------------------------------
// 3) bf16 WMMA GEMM:  D[M,N] = A[M,K] * B[N,K]^T (+ bias[n])
//    One wave per 16x16 tile; v_wmma_f32_16x16x32_bf16 with f32 accumulator.
//    A per-lane layout per ISA (K 0-7 & 16-23 for lanes 0-15; 8-15 & 24-31
//    for lanes 16-31); B per-lane: same K split as A.
// ---------------------------------------------------------------------------
__global__ void wmma_gemm_bf16(const unsigned short* __restrict__ A,
                               const unsigned short* __restrict__ Bm,
                               float* __restrict__ D,
                               const float* __restrict__ bias,
                               int M, int N, int K) {
    int wave = (blockIdx.x * blockDim.x + threadIdx.x) >> 5;
    int lane = threadIdx.x & 31;
    int ntn  = N >> 4;
    int tm = wave / ntn, tn = wave - tm * ntn;
    if (tm >= (M >> 4)) return;

    int half = lane >> 4;       // 0 | 1
    int l15  = lane & 15;
    const unsigned short* arow = A  + (size_t)(tm * 16 + l15) * K;
    const unsigned short* brow = Bm + (size_t)(tn * 16 + l15) * K;
    const int abase = half * 8;      // A: two 8-elem chunks, 16 apart
    const int bbase = half * 8;      // B: two 8-elem chunks, 16 apart

    v8f acc = {};
#pragma unroll 4
    for (int k0 = 0; k0 < K; k0 += 32) {
        union { v16bf v; uint4 u[2]; } a, b;
        a.u[0] = *(const uint4*)(arow + k0 + abase);
        a.u[1] = *(const uint4*)(arow + k0 + abase + 16);
        b.u[0] = *(const uint4*)(brow + k0 + bbase);
        b.u[1] = *(const uint4*)(brow + k0 + bbase + 16);
        acc = __builtin_amdgcn_wmma_f32_16x16x32_bf16(
            false, a.v, false, b.v, (short)0, acc, false, false);
    }

    int col  = tn * 16 + l15;
    float bs = bias ? bias[col] : 0.f;
    int row0 = tm * 16 + half * 8;     // D layout: VGPR r -> M = row0 + r
#pragma unroll
    for (int r = 0; r < 8; ++r)
        D[(size_t)(row0 + r) * N + col] = acc[r] + bs;
}

// ---------------------------------------------------------------------------
// 4) Attention, last query row only. One wave per (b, head).
//    rel bias for query row 8: rel = clip(8-t,-8,8)+8 = 16-t.
// ---------------------------------------------------------------------------
__global__ void attn_kernel(const float* __restrict__ qkv,
                            const float* __restrict__ rel_emb,
                            unsigned short* __restrict__ outvec) {
    int wave = (blockIdx.x * blockDim.x + threadIdx.x) >> 5;   // 0..255
    int lane = threadIdx.x & 31;
    int b = wave >> 3, h = wave & 7;
    const float* base = qkv + (size_t)b * T_SZ * (3 * HIDDEN);

    float4 q = ((const float4*)(base + (T_SZ - 1) * 3 * HIDDEN + h * HD))[lane];
    const float scale = 0.08838834764831845f;   // 1/sqrt(128)

    float sc[T_SZ];
#pragma unroll
    for (int t = 0; t < T_SZ; ++t) {
        float4 k = ((const float4*)(base + t * 3 * HIDDEN + HIDDEN + h * HD))[lane];
        float s = q.x * k.x + q.y * k.y + q.z * k.z + q.w * k.w;
#pragma unroll
        for (int off = 16; off; off >>= 1) s += __shfl_xor(s, off, 32);
        sc[t] = s * scale + rel_emb[(16 - t) * NHEAD + h];
    }
    float mx = sc[0];
#pragma unroll
    for (int t = 1; t < T_SZ; ++t) mx = fmaxf(mx, sc[t]);
    float sum = 0.f;
#pragma unroll
    for (int t = 0; t < T_SZ; ++t) { sc[t] = __expf(sc[t] - mx); sum += sc[t]; }
    float inv = 1.0f / sum;

    float4 o = {0.f, 0.f, 0.f, 0.f};
#pragma unroll
    for (int t = 0; t < T_SZ; ++t) {
        float4 v = ((const float4*)(base + t * 3 * HIDDEN + 2 * HIDDEN + h * HD))[lane];
        float p = sc[t] * inv;
        o.x += p * v.x; o.y += p * v.y; o.z += p * v.z; o.w += p * v.w;
    }
    unsigned short* dst = outvec + (size_t)b * HIDDEN + h * HD + lane * 4;
    dst[0] = f2bf(o.x); dst[1] = f2bf(o.y); dst[2] = f2bf(o.z); dst[3] = f2bf(o.w);
}

// ---------------------------------------------------------------------------
// 5) Residual broadcast: out[b,c,h,w] = h_current[b,c,h,w] + proj[b,c]
//    Pure stream (75 MB in + 75 MB out): NT load + NT store; proj stays
//    cached (regular load, 128 KB, heavily reused).
// ---------------------------------------------------------------------------
__global__ void residual_kernel(const float* __restrict__ hc,
                                const float* __restrict__ proj,
                                float* __restrict__ outp, int n4) {
    int i = blockIdx.x * blockDim.x + threadIdx.x;
    if (i >= n4) return;
    float add = proj[i / (SPATIAL / 4)];
    fvec4 v = __builtin_nontemporal_load((const fvec4*)hc + i);
    v.x += add; v.y += add; v.z += add; v.w += add;
    __builtin_nontemporal_store(v, (fvec4*)outp + i);
}

// ---------------------------------------------------------------------------
extern "C" void kernel_launch(void* const* d_in, const int* in_sizes, int n_in,
                              void* d_out, int out_size, void* d_ws, size_t ws_size,
                              hipStream_t stream) {
    (void)in_sizes; (void)n_in; (void)out_size; (void)ws_size;
    const float* h_current = (const float*)d_in[0];
    const float* h_buffer  = (const float*)d_in[1];
    const float* W_qkv     = (const float*)d_in[2];
    const float* W_out     = (const float*)d_in[3];
    const float* b_out     = (const float*)d_in[4];
    const float* rel_emb   = (const float*)d_in[5];
    float* out = (float*)d_out;

    char* ws = (char*)d_ws;
    unsigned short* ctx_bf    = (unsigned short*)ws; ws += (size_t)B_SZ * T_SZ * HIDDEN * 2;      // 576 KB
    unsigned short* wqkv_bf   = (unsigned short*)ws; ws += (size_t)3 * HIDDEN * HIDDEN * 2;       // 6 MB
    unsigned short* wout_bf   = (unsigned short*)ws; ws += (size_t)HIDDEN * HIDDEN * 2;           // 2 MB
    float*          qkv       = (float*)ws;          ws += (size_t)B_SZ * T_SZ * 3 * HIDDEN * 4;  // 3.4 MB
    unsigned short* outvec_bf = (unsigned short*)ws; ws += (size_t)B_SZ * HIDDEN * 2;             // 64 KB
    float*          proj      = (float*)ws;          ws += (size_t)B_SZ * HIDDEN * 4;             // 128 KB

    // 1) pooling: 294912 waves = 36864 blocks x 8 waves
    pool_kernel<<<36864, 256, 0, stream>>>(h_current, h_buffer, ctx_bf);
    // 2) weight conversion
    convert_kernel<<<(3 * HIDDEN * HIDDEN) / 256, 256, 0, stream>>>(W_qkv, wqkv_bf, 3 * HIDDEN * HIDDEN);
    convert_kernel<<<(HIDDEN * HIDDEN) / 256, 256, 0, stream>>>(W_out, wout_bf, HIDDEN * HIDDEN);
    // 3) QKV GEMM: [288 x 3072 x 1024] -> 18*192 = 3456 tiles -> 432 blocks
    wmma_gemm_bf16<<<432, 256, 0, stream>>>(ctx_bf, wqkv_bf, qkv, nullptr,
                                            B_SZ * T_SZ, 3 * HIDDEN, HIDDEN);
    // 4) attention (last query row): 256 waves -> 32 blocks
    attn_kernel<<<32, 256, 0, stream>>>(qkv, rel_emb, outvec_bf);
    // 5) out-projection with bias: [32 x 1024 x 1024] -> 128 tiles -> 16 blocks
    wmma_gemm_bf16<<<16, 256, 0, stream>>>(outvec_bf, wout_bf, proj, b_out,
                                           B_SZ, HIDDEN, HIDDEN);
    // 6) residual broadcast: 4,718,592 x 16B
    residual_kernel<<<18432, 256, 0, stream>>>(h_current, proj, out,
                                               (B_SZ * HIDDEN * SPATIAL) / 4);
}
